// KANLinear_42949673133
// MI455X (gfx1250) — matrix operand order repaired and checked by reading, not compiled, not run on hardware
//
#include <hip/hip_runtime.h>
#include <math.h>

typedef __attribute__((ext_vector_type(16))) _Float16 v16h;
typedef __attribute__((ext_vector_type(8)))  _Float16 v8h;
typedef __attribute__((ext_vector_type(8)))  float    v8f;
typedef __attribute__((ext_vector_type(4)))  int      v4i;

// Problem constants (from reference)
#define NTOK   16384
#define INF    128
#define OUTF   128
#define KSPL   8
#define KTOT   1152          // 128 silu + 1024 spline-basis
#define NKT    36            // KTOT / 32
#define MROWS  64
#define LDSTRIDE 1160        // act row stride in halves (2320B -> conflict-free ds_load_b128)
#define ACT_BYTES (MROWS * LDSTRIDE * 2)          // 148480

#define CHUNK_KT 9
#define NCHUNK   (NKT / CHUNK_KT)                 // 4
#define FRAG_B   1024                             // bytes per B fragment (32 lanes x 32B)
#define FRAG_STRIDE 1040                          // +16B pad in LDS
#define NFRAG_CHUNK (CHUNK_KT * 8)                // 72 fragments per chunk
#define BBUF_BYTES (NFRAG_CHUNK * FRAG_STRIDE)    // 74880
#define CHUNK_GBYTES (NFRAG_CHUNK * FRAG_B)       // 73728 contiguous in wpk
#define XFERS (CHUNK_GBYTES / 16)                 // 4608 b128 transfers / chunk (18/thread)
#define TOTAL_LDS (ACT_BYTES + 2 * BBUF_BYTES)    // 298240 <= 320KB/WGP

// ---- CDNA5 async global->LDS path (guarded; sync fallback keeps compile OK) ----
#if __has_builtin(__builtin_amdgcn_global_load_async_to_lds_b128)
#define HAVE_ASYNC 1
#define ASYNC_COPY_B128(gsrc, ldst)                                            \
    __builtin_amdgcn_global_load_async_to_lds_b128(                            \
        (__attribute__((address_space(1))) v4i*)(gsrc),                        \
        (__attribute__((address_space(3))) v4i*)(ldst), 0, 0)
#else
#define HAVE_ASYNC 0
#define ASYNC_COPY_B128(gsrc, ldst)                                            \
    do { *(v8h*)(ldst) = *(const v8h*)(gsrc); } while (0)
#endif

#if __has_builtin(__builtin_amdgcn_s_wait_asynccnt)
#define WAIT_ASYNC0() __builtin_amdgcn_s_wait_asynccnt(0)
#elif HAVE_ASYNC
#define WAIT_ASYNC0() asm volatile("s_wait_asynccnt 0x0" ::: "memory")
#else
#define WAIT_ASYNC0() do { } while (0)
#endif

// Scheduling-group hints: pin software pipelining of DS-reads vs WMMAs.
// Masks: 0x100 = DS read, 0x8 = MFMA/WMMA group.
#if __has_builtin(__builtin_amdgcn_sched_group_barrier)
#define SGB(mask, n) __builtin_amdgcn_sched_group_barrier((mask), (n), 0)
#else
#define SGB(mask, n) do { } while (0)
#endif

// ---------------------------------------------------------------------------
// Pre-pass: pack merged weight [base | spline*scaler] (128x1152) into f16
// WMMA B-fragment layout, K-major so each K-chunk is one contiguous block:
//   wpk[((kt*8 + nt)*32 + lane)*16 + e] = W[n][K]
//   n = nt*16 + (lane&15),  K = kt*32 + (lane&16 ? 16 : 0) + e
// ---------------------------------------------------------------------------
__global__ void __launch_bounds__(256)
kan_pack_w(const float* __restrict__ bw, const float* __restrict__ sw,
           const float* __restrict__ sc, _Float16* __restrict__ wpk) {
    int gid  = blockIdx.x * 256 + threadIdx.x;   // 0 .. 36*8*32-1
    int lane = gid & 31;
    int frag = gid >> 5;                         // kt*8 + nt
    int nt   = frag & 7;
    int kt   = frag >> 3;
    int n     = nt * 16 + (lane & 15);
    int kbase = kt * 32 + ((lane & 16) ? 16 : 0);
    v16h v;
#pragma unroll
    for (int e = 0; e < 16; ++e) {
        int K = kbase + e;
        float w;
        if (K < INF) {
            w = bw[n * INF + K];
        } else {
            int r = K - INF;
            int i = r >> 3, k = r & 7;
            w = sw[(n * INF + i) * KSPL + k] * sc[n * INF + i];
        }
        v[e] = (_Float16)w;
    }
    *(v16h*)(wpk + (size_t)gid * 16) = v;
}

// ---------------------------------------------------------------------------
// Main fused kernel
// ---------------------------------------------------------------------------
__device__ __forceinline__ void issue_chunk_copy(const _Float16* wpk, char* smem,
                                                 int c, int tid) {
    const char* gsrc = (const char*)wpk + (size_t)c * CHUNK_GBYTES;
    char* lbase = smem + ACT_BYTES + (size_t)(c & 1) * BBUF_BYTES;
#pragma unroll
    for (int t = tid; t < XFERS; t += 256) {
        int f    = t >> 6;                            // fragment index in chunk
        int u    = t & 63;                            // 16B unit in fragment
        int lfr  = u >> 1;                            // owning lane
        int half = (u & 1) ^ ((lfr >> 3) & 1);        // bank-conflict swizzle
        ASYNC_COPY_B128(gsrc + (size_t)t * 16,
                        lbase + f * FRAG_STRIDE + lfr * 32 + half * 16);
    }
}

struct Frags { v16h a0, a1, b0, b1; };

__device__ __forceinline__ v16h ld_a_frag(const _Float16* ap) {
    v16h a;
    *(v8h*)&a         = *(const v8h*)(ap);
    *(((v8h*)&a) + 1) = *(const v8h*)(ap + 8);
    return a;
}
__device__ __forceinline__ v16h ld_b_frag(const char* fragbase, int lane) {
    const int hsel = ((lane >> 3) & 1) << 4;          // undo copy-side swizzle
    const char* p = fragbase + lane * 32;
    v16h b;
    *(v8h*)&b         = *(const v8h*)(p + hsel);
    *(((v8h*)&b) + 1) = *(const v8h*)(p + (16 - hsel));
    return b;
}

__device__ __forceinline__ Frags ld_frags(const _Float16* act, const char* lbase,
                                          int arow, int aoff, int nt0, int kt, int ktl,
                                          int lane) {
    Frags f;
    const _Float16* ap = &act[arow + kt * 32 + aoff];
    f.a0 = ld_a_frag(ap);
    f.a1 = ld_a_frag(ap + 16 * LDSTRIDE);
    const char* bl = lbase + (ktl * 8 + nt0) * FRAG_STRIDE;
    f.b0 = ld_b_frag(bl, lane);
    f.b1 = ld_b_frag(bl + FRAG_STRIDE, lane);
    return f;
}

__device__ __forceinline__ void do_wmma(const Frags& f, v8f& c00, v8f& c01,
                                        v8f& c10, v8f& c11) {
    c00 = __builtin_amdgcn_wmma_f32_16x16x32_f16(false, f.a0, false, f.b0, (short)0, c00, false, false);
    c01 = __builtin_amdgcn_wmma_f32_16x16x32_f16(false, f.a0, false, f.b1, (short)0, c01, false, false);
    c10 = __builtin_amdgcn_wmma_f32_16x16x32_f16(false, f.a1, false, f.b0, (short)0, c10, false, false);
    c11 = __builtin_amdgcn_wmma_f32_16x16x32_f16(false, f.a1, false, f.b1, (short)0, c11, false, false);
}

__global__ void __launch_bounds__(256)
kan_main(const float* __restrict__ X, const _Float16* __restrict__ wpk,
         float* __restrict__ out) {
    extern __shared__ char smem[];
    _Float16* act = (_Float16*)smem;
    const int tid = threadIdx.x;
    const int m0  = blockIdx.x * MROWS;

    // Kick chunk-0 B copy immediately: overlaps with activation build.
    issue_chunk_copy(wpk, smem, 0, tid);

    // ---- Phase 1a: silu(x) -> act[row][0..127] -------------------------
    for (int idx = tid; idx < MROWS * INF; idx += 256) {
        int row = idx >> 7, c = idx & 127;
        float x = X[(size_t)(m0 + row) * INF + c];
        float s = x / (1.0f + __expf(-x));
        act[row * LDSTRIDE + c] = (_Float16)s;
    }

    // ---- Phase 1b: cubic B-spline basis (uniform grid, closed form) ----
    for (int idx = tid; idx < MROWS * INF; idx += 256) {
        int row = idx >> 7, i = idx & 127;
        float x  = X[(size_t)(m0 + row) * INF + i];
        float t  = (x + 1.0f) * 2.5f;            // (x+1)/h, h=0.4
        float fj = floorf(t);
        float u  = t - fj;
        int   j  = (int)fj;
        j = j < 0 ? 0 : (j > 4 ? 4 : j);
        float u2 = u * u, u3 = u2 * u;
        float w0 = (1.0f - u);  w0 = w0 * w0 * w0 * (1.0f / 6.0f);
        float w1 = (3.0f * u3 - 6.0f * u2 + 4.0f) * (1.0f / 6.0f);
        float w2 = (-3.0f * u3 + 3.0f * u2 + 3.0f * u + 1.0f) * (1.0f / 6.0f);
        float w3 = u3 * (1.0f / 6.0f);
        v8h frag;
#pragma unroll
        for (int k = 0; k < 8; ++k) {
            float v = 0.0f;
            v = (k == j    ) ? w0 : v;
            v = (k == j + 1) ? w1 : v;
            v = (k == j + 2) ? w2 : v;
            v = (k == j + 3) ? w3 : v;
            frag[k] = (_Float16)v;
        }
        *(v8h*)&act[row * LDSTRIDE + INF + i * 8] = frag;
    }

    // ---- Phase 2: WMMA GEMM, B double-buffered in LDS via async copy ----
    // Wave (mg,nh): 2 m-tiles {2mg,2mg+1} x 2 n-tiles {2nh,2nh+1}
    const int wave = tid >> 5;
    const int lane = tid & 31;
    const int mg   = wave >> 2;                   // 0..1
    const int nh   = wave & 3;                    // 0..3
    const int arow = ((mg << 5) + (lane & 15)) * LDSTRIDE;
    const int aoff = (lane & 16) ? 16 : 0;
    const int nt0  = nh << 1;

    v8f c00 = {}, c01 = {}, c10 = {}, c11 = {};

    for (int c = 0; c < NCHUNK; ++c) {
        WAIT_ASYNC0();            // this wave's chunk-c transfers are in LDS
        __syncthreads();          // all waves' transfers visible; prev compute done
        if (c + 1 < NCHUNK) issue_chunk_copy(wpk, smem, c + 1, tid);

        const char* lbase = smem + ACT_BYTES + (size_t)(c & 1) * BBUF_BYTES;
        // Pipelined chunk body. Group sequence (after full unroll):
        //   DS(16), [WMMA(4), DS(8)] x7, WMMA(4), WMMA(4)
        // => iteration k+1's 8 LDS loads are scheduled ahead of iteration k's
        //    WMMAs, so waits keep ~8 loads in flight instead of draining.
#pragma unroll
        for (int ktl = 0; ktl < CHUNK_KT; ++ktl) {
            Frags f = ld_frags(act, lbase, arow, aoff, nt0,
                               c * CHUNK_KT + ktl, ktl, lane);
            do_wmma(f, c00, c01, c10, c11);
            if (ktl == 0)                 { SGB(0x100, 16); }
            else if (ktl < CHUNK_KT - 1)  { SGB(0x100, 8);  }
            SGB(0x8, 4);
        }
    }

    // ---- Epilogue: f32 C/D layout (8 VGPRs, lanes 16-31 = M+8) ---------
    const int orow0 = m0 + (mg << 5) + ((lane & 16) ? 8 : 0);
    const int ocol  = (nt0 << 4) + (lane & 15);
#pragma unroll
    for (int r = 0; r < 8; ++r) {
        float* o0 = out + (size_t)(orow0 + r) * OUTF + ocol;
        float* o1 = out + (size_t)(orow0 + 16 + r) * OUTF + ocol;
        o0[0]  = c00[r];
        o0[16] = c01[r];
        o1[0]  = c10[r];
        o1[16] = c11[r];
    }
}

extern "C" void kernel_launch(void* const* d_in, const int* in_sizes, int n_in,
                              void* d_out, int out_size, void* d_ws, size_t ws_size,
                              hipStream_t stream) {
    const float* x  = (const float*)d_in[0];   // (16384,128)
    const float* bw = (const float*)d_in[1];   // (128,128)
    const float* sw = (const float*)d_in[2];   // (128,128,8)
    const float* sc = (const float*)d_in[3];   // (128,128)
    // d_in[4] = grid: uniform, recomputed in closed form on device

    _Float16* wpk = (_Float16*)d_ws;           // 36*8*32*16 halves = 288 KB

    kan_pack_w<<<NKT, 256, 0, stream>>>(bw, sw, sc, wpk);
    kan_main<<<NTOK / MROWS, 256, TOTAL_LDS, stream>>>(x, wpk, (float*)d_out);
}
